// MoEBlock_39745627357692
// MI455X (gfx1250) — matrix operand order repaired and checked by reading, not compiled, not run on hardware
//
#include <hip/hip_runtime.h>
#include <hip/hip_bf16.h>
#include <math.h>

// ---------------------------------------------------------------------------
// MoE block (top-2 of 8 experts), B=4 T=2048 C=1024 H=4096, fp32 in/out.
// Strategy: bf16 WMMA (v_wmma_f32_16x16x32_bf16), fused 2-GEMM FFN with the
// 32x4096 hidden activation staged in 263KB of LDS (CDNA5: 320KB/WGP).
// Inner loops: 2x2 register blocking, limited unroll, per-n-block compiler
// memory barrier (prevents LICM scratch spills of A fragments).
// Epilogue: branch-free erf (A&S 7.1.26, |err|<1.5e-7) with hardware
// v_rcp_f32 / v_exp_f32 instead of the divergent ocml erff.
// ---------------------------------------------------------------------------

#define C_DIM 1024
#define H_DIM 4096
#define E_NUM 8
#define N_TOK 8192
#define M_TILE 32
#define HPAD (H_DIM + 8)                  // LDS row pad -> 16B-aligned, conflict-free rows
#define TILES_PER_E ((N_TOK + M_TILE - 1) / M_TILE)   // 256
#define SHMEM_BYTES (M_TILE * HPAD * 2 + M_TILE * 4 + M_TILE * 4)

typedef __attribute__((ext_vector_type(16))) __bf16 v16bf;
typedef __attribute__((ext_vector_type(8)))  float  v8f;

static __device__ __forceinline__ unsigned short f32_to_bf16(float f) {
  union { float f; unsigned u; } v; v.f = f;
  unsigned r = v.u + 0x7FFFu + ((v.u >> 16) & 1u);   // round-to-nearest-even
  return (unsigned short)(r >> 16);
}

union Frag32 { uint4 q[2]; v16bf v; };

// A fragment (16x32 bf16, row-major source, per ISA 16-bit A layout):
// lane<16: K 0..7 then 16..23 ; lane>=16: K 8..15 then 24..31.
// p points at (row, k0 + koff) where koff = (lane>>4)*8. Two 16B loads.
static __device__ __forceinline__ v16bf load_a_frag(const unsigned short* p) {
  Frag32 f;
  f.q[0] = *(const uint4*)(p);
  f.q[1] = *(const uint4*)(p + 16);
  return f.v;
}

// B fragment (32x16 bf16) from K-contiguous (transposed-weight) row:
// lane<16: K 0..7 then 16..23 ; lane>=16: K 8..15 then 24..31. p points at (ncol-row, k0 + bko),
// bko = (lane>>4)*8. Two 16B loads.
static __device__ __forceinline__ v16bf load_b_frag(const unsigned short* p) {
  Frag32 f;
  f.q[0] = *(const uint4*)(p);
  f.q[1] = *(const uint4*)(p + 16);
  return f.v;
}

static __device__ __forceinline__ v8f wmma_bf16(v16bf a, v16bf b, v8f c) {
  return __builtin_amdgcn_wmma_f32_16x16x32_bf16(false, a, false, b, (short)0, c,
                                                 false, false);
}

// Branch-free GELU(exact-erf form): erf via Abramowitz-Stegun 7.1.26
// (|error| <= 1.5e-7, far below bf16 resolution), hardware rcp + exp.
static __device__ __forceinline__ float gelu_fast(float v) {
  float s  = v * 0.70710678118654752f;          // v / sqrt(2)
  float ax = __builtin_fabsf(s);
  float t  = __builtin_amdgcn_rcpf(__builtin_fmaf(0.3275911f, ax, 1.0f));
  float p  = __builtin_fmaf(t, 1.061405429f, -1.453152027f);
  p = __builtin_fmaf(t, p, 1.421413741f);
  p = __builtin_fmaf(t, p, -0.284496736f);
  p = __builtin_fmaf(t, p, 0.254829592f);
  p = p * t;
  float e  = __expf(-ax * ax);                  // v_exp_f32 path
  float ea = __builtin_fmaf(-p, e, 1.0f);       // erf(|s|)
  float er = (s < 0.0f) ? -ea : ea;             // v_cndmask, no divergence
  return 0.5f * v * (1.0f + er);
}

// ---------------------------------------------------------------------------
// 0) zero output + expert counters
// ---------------------------------------------------------------------------
__global__ void zero_init_kernel(float* __restrict__ out, size_t n,
                                 int* __restrict__ cnt) {
  size_t i = (size_t)blockIdx.x * blockDim.x + threadIdx.x;
  if (i < n) out[i] = 0.0f;
  if (i < E_NUM) cnt[i] = 0;
}

// ---------------------------------------------------------------------------
// 1) x (fp32) -> bf16
// ---------------------------------------------------------------------------
__global__ void cvt_x_kernel(const float* __restrict__ x,
                             unsigned short* __restrict__ xbf, size_t n) {
  size_t i = (size_t)blockIdx.x * blockDim.x + threadIdx.x;
  if (i < n) xbf[i] = f32_to_bf16(x[i]);
}

// ---------------------------------------------------------------------------
// 2) tiled transpose + fp32->bf16: src[e][rows][cols] -> dst[e][cols][rows]
//    block (32,8), grid (cols/32, rows/32, E)
// ---------------------------------------------------------------------------
__global__ void transpose_bf16_kernel(const float* __restrict__ src,
                                      unsigned short* __restrict__ dst,
                                      int rows, int cols) {
  __shared__ float tile[32][33];
  const size_t plane = (size_t)rows * cols;
  const float* s = src + (size_t)blockIdx.z * plane;
  unsigned short* d = dst + (size_t)blockIdx.z * plane;
  int c0 = blockIdx.x * 32;
  int r0 = blockIdx.y * 32;
  int tx = threadIdx.x, ty = threadIdx.y;
#pragma unroll
  for (int j = 0; j < 32; j += 8)
    tile[ty + j][tx] = s[(size_t)(r0 + ty + j) * cols + (c0 + tx)];
  __syncthreads();
#pragma unroll
  for (int j = 0; j < 32; j += 8)
    d[(size_t)(c0 + ty + j) * rows + (r0 + tx)] = f32_to_bf16(tile[tx][ty + j]);
}

// ---------------------------------------------------------------------------
// 3) gating: one wave32 per token. logits = x_row @ w_gate (C x 8),
//    top-2 + softmax, scatter (token, weight) into per-expert lists.
// ---------------------------------------------------------------------------
__global__ void gate_kernel(const float* __restrict__ x,
                            const float* __restrict__ wg,
                            int* __restrict__ cnt,
                            int* __restrict__ tokidx,
                            float* __restrict__ tokw) {
  int lane = threadIdx.x & 31;
  int wave = threadIdx.x >> 5;
  int token = blockIdx.x * (blockDim.x >> 5) + wave;
  if (token >= N_TOK) return;
  const float* xr = x + (size_t)token * C_DIM;
  float acc[E_NUM] = {0.f, 0.f, 0.f, 0.f, 0.f, 0.f, 0.f, 0.f};
  for (int c = lane; c < C_DIM; c += 32) {
    float xv = xr[c];
    const float4* wr = (const float4*)(wg + (size_t)c * E_NUM);
    float4 w0 = wr[0], w1v = wr[1];
    acc[0] += xv * w0.x;  acc[1] += xv * w0.y;
    acc[2] += xv * w0.z;  acc[3] += xv * w0.w;
    acc[4] += xv * w1v.x; acc[5] += xv * w1v.y;
    acc[6] += xv * w1v.z; acc[7] += xv * w1v.w;
  }
#pragma unroll
  for (int e = 0; e < E_NUM; ++e)
#pragma unroll
    for (int off = 16; off > 0; off >>= 1)
      acc[e] += __shfl_down(acc[e], off, 32);
  if (lane == 0) {
    float v1 = -1e30f, v2 = -1e30f; int i1 = 0, i2 = 0;
#pragma unroll
    for (int e = 0; e < E_NUM; ++e) {
      float v = acc[e];
      if (v > v1)      { v2 = v1; i2 = i1; v1 = v; i1 = e; }
      else if (v > v2) { v2 = v; i2 = e; }
    }
    float s1 = 1.0f / (1.0f + __expf(v2 - v1));   // softmax over {v1, v2}
    float s2 = 1.0f - s1;
    int slot1 = atomicAdd(&cnt[i1], 1);
    tokidx[i1 * N_TOK + slot1] = token; tokw[i1 * N_TOK + slot1] = s1;
    int slot2 = atomicAdd(&cnt[i2], 1);
    tokidx[i2 * N_TOK + slot2] = token; tokw[i2 * N_TOK + slot2] = s2;
  }
}

// ---------------------------------------------------------------------------
// 4) fused expert FFN: one WG = (expert, 32-token tile), 256 thr = 8 waves.
//    GEMM1: H = gelu(X @ W1 + b1) -> LDS (32 x 4096 bf16, padded rows)
//    GEMM2: Y = H @ W2 + b2 ; out[token] += route_w * Y   (atomic f32)
// ---------------------------------------------------------------------------
__global__ void moe_ffn_kernel(const unsigned short* __restrict__ xbf,   // [N][C]
                               const unsigned short* __restrict__ w1t,   // [E][H][C]
                               const float* __restrict__ b1,             // [E][H]
                               const unsigned short* __restrict__ w2t,   // [E][C][H]
                               const float* __restrict__ b2,             // [E][C]
                               const int* __restrict__ cnt,
                               const int* __restrict__ tokidx,
                               const float* __restrict__ tokw,
                               float* __restrict__ out) {
  extern __shared__ char smem[];
  unsigned short* Hbuf = (unsigned short*)smem;             // [M_TILE][HPAD]
  int*   tokS = (int*)(smem + (size_t)M_TILE * HPAD * 2);
  float* wtS  = (float*)(tokS + M_TILE);

  const int e    = blockIdx.x >> 8;        // / TILES_PER_E (=256)
  const int tile = blockIdx.x & 255;
  const int count = cnt[e];
  if (tile * M_TILE >= count) return;      // block-uniform early exit

  const int tid = threadIdx.x;
  const int lane = tid & 31;
  const int wave = tid >> 5;

  if (tid < M_TILE) {
    int slot = tile * M_TILE + tid;
    if (slot < count) {
      tokS[tid] = tokidx[e * N_TOK + slot];
      wtS[tid]  = tokw[e * N_TOK + slot];
    } else {
      tokS[tid] = 0;                        // padded slot: weight 0
      wtS[tid]  = 0.0f;
    }
  }
  __syncthreads();

  const int mrow = lane & 15;
  const int koff = (lane >> 4) * 8;         // A-fragment K sub-offset
  const int bko  = (lane >> 4) * 8;         // B-fragment K sub-offset
  const int msub = (lane >> 4) << 3;        // C/D row base per half-wave
  const unsigned short* arow0 = xbf + (size_t)tokS[mrow] * C_DIM;
  const unsigned short* arow1 = xbf + (size_t)tokS[mrow + 16] * C_DIM;

  // ---------------- GEMM1: 32x4096 = X(32x1024) @ W1(1024x4096) -------------
  const unsigned short* W1e = w1t + (size_t)e * H_DIM * C_DIM;
  const float* b1e = b1 + e * H_DIM;
  for (int np = 0; np < 16; ++np) {         // 8 waves * 16 pairs * 32 = 4096 cols
    asm volatile("" ::: "memory");          // block cross-n hoist of A loads
    const int n0 = wave * 512 + np * 32 + (lane & 15);
    const int n1 = n0 + 16;
    const unsigned short* brow0 = W1e + (size_t)n0 * C_DIM;
    const unsigned short* brow1 = W1e + (size_t)n1 * C_DIM;
    v8f acc00 = {}; v8f acc01 = {}; v8f acc10 = {}; v8f acc11 = {};
#pragma clang loop unroll_count(2)
    for (int k0 = 0; k0 < C_DIM; k0 += 32) {
      v16bf a0 = load_a_frag(arow0 + k0 + koff);
      v16bf a1 = load_a_frag(arow1 + k0 + koff);
      v16bf b0 = load_b_frag(brow0 + k0 + bko);
      v16bf b1f = load_b_frag(brow1 + k0 + bko);
      acc00 = wmma_bf16(a0, b0, acc00);
      acc10 = wmma_bf16(a1, b0, acc10);
      acc01 = wmma_bf16(a0, b1f, acc01);
      acc11 = wmma_bf16(a1, b1f, acc11);
    }
    const float bias0 = b1e[n0];
    const float bias1 = b1e[n1];
#pragma unroll
    for (int r = 0; r < 8; ++r) {
      int m0 = r + msub;                              // rows 0..15
      Hbuf[(size_t)m0 * HPAD + n0]        = f32_to_bf16(gelu_fast(acc00[r] + bias0));
      Hbuf[(size_t)(m0 + 16) * HPAD + n0] = f32_to_bf16(gelu_fast(acc10[r] + bias0));
      Hbuf[(size_t)m0 * HPAD + n1]        = f32_to_bf16(gelu_fast(acc01[r] + bias1));
      Hbuf[(size_t)(m0 + 16) * HPAD + n1] = f32_to_bf16(gelu_fast(acc11[r] + bias1));
    }
  }
  __syncthreads();

  // ---------------- GEMM2: 32x1024 = H(32x4096) @ W2(4096x1024) -------------
  const unsigned short* W2e = w2t + (size_t)e * C_DIM * H_DIM;
  const float* b2e = b2 + e * C_DIM;
  const unsigned short* hrow0 = Hbuf + (size_t)mrow * HPAD;
  const unsigned short* hrow1 = Hbuf + (size_t)(mrow + 16) * HPAD;
  for (int np = 0; np < 4; ++np) {          // 8 waves * 4 pairs * 32 = 1024 cols
    asm volatile("" ::: "memory");          // block cross-n hoist of A loads
    const int n0 = wave * 128 + np * 32 + (lane & 15);
    const int n1 = n0 + 16;
    const unsigned short* brow0 = W2e + (size_t)n0 * H_DIM;
    const unsigned short* brow1 = W2e + (size_t)n1 * H_DIM;
    v8f acc00 = {}; v8f acc01 = {}; v8f acc10 = {}; v8f acc11 = {};
#pragma clang loop unroll_count(2)
    for (int k0 = 0; k0 < H_DIM; k0 += 32) {
      v16bf a0 = load_a_frag(hrow0 + k0 + koff);      // LDS
      v16bf a1 = load_a_frag(hrow1 + k0 + koff);      // LDS
      v16bf b0 = load_b_frag(brow0 + k0 + bko);       // global
      v16bf b1f = load_b_frag(brow1 + k0 + bko);      // global
      acc00 = wmma_bf16(a0, b0, acc00);
      acc10 = wmma_bf16(a1, b0, acc10);
      acc01 = wmma_bf16(a0, b1f, acc01);
      acc11 = wmma_bf16(a1, b1f, acc11);
    }
    const float bias0 = b2e[n0];
    const float bias1 = b2e[n1];
#pragma unroll
    for (int r = 0; r < 8; ++r) {
      int m0 = r + msub;
      int m1 = m0 + 16;
      float y;
      y = (acc00[r] + bias0) * wtS[m0];
      atomicAdd(&out[(size_t)tokS[m0] * C_DIM + n0], y);
      y = (acc10[r] + bias0) * wtS[m1];
      atomicAdd(&out[(size_t)tokS[m1] * C_DIM + n0], y);
      y = (acc01[r] + bias1) * wtS[m0];
      atomicAdd(&out[(size_t)tokS[m0] * C_DIM + n1], y);
      y = (acc11[r] + bias1) * wtS[m1];
      atomicAdd(&out[(size_t)tokS[m1] * C_DIM + n1], y);
    }
  }
}

// ---------------------------------------------------------------------------
// host launcher
// ---------------------------------------------------------------------------
extern "C" void kernel_launch(void* const* d_in, const int* in_sizes, int n_in,
                              void* d_out, int out_size, void* d_ws, size_t ws_size,
                              hipStream_t stream) {
  (void)in_sizes; (void)n_in; (void)out_size; (void)ws_size;
  const float* x  = (const float*)d_in[0];   // [4,2048,1024]
  const float* wg = (const float*)d_in[1];   // [1024,8]
  const float* w1 = (const float*)d_in[2];   // [8,1024,4096]
  const float* b1 = (const float*)d_in[3];   // [8,4096]
  const float* w2 = (const float*)d_in[4];   // [8,4096,1024]
  const float* b2 = (const float*)d_in[5];   // [8,1024]
  float* out = (float*)d_out;                // [4,2048,1024]

  // workspace carve-up
  char* ws = (char*)d_ws;
  unsigned short* xbf = (unsigned short*)ws;  ws += (size_t)N_TOK * C_DIM * 2;         // 16 MB
  unsigned short* w1t = (unsigned short*)ws;  ws += (size_t)E_NUM * H_DIM * C_DIM * 2; // 67 MB
  unsigned short* w2t = (unsigned short*)ws;  ws += (size_t)E_NUM * C_DIM * H_DIM * 2; // 67 MB
  int*   cnt    = (int*)ws;                   ws += 256;
  int*   tokidx = (int*)ws;                   ws += (size_t)E_NUM * N_TOK * 4;
  float* tokw   = (float*)ws;                 ws += (size_t)E_NUM * N_TOK * 4;

  const size_t n_out = (size_t)N_TOK * C_DIM;

  // 0) zero out + counters
  zero_init_kernel<<<(int)((n_out + 255) / 256), 256, 0, stream>>>(out, n_out, cnt);

  // 1) x -> bf16
  cvt_x_kernel<<<(int)((n_out + 255) / 256), 256, 0, stream>>>(x, xbf, n_out);

  // 2) weight transposes to K-contiguous bf16
  {
    dim3 blk(32, 8);
    dim3 g1(H_DIM / 32, C_DIM / 32, E_NUM);  // w1[e][C][H] -> w1t[e][H][C]
    transpose_bf16_kernel<<<g1, blk, 0, stream>>>(w1, w1t, C_DIM, H_DIM);
    dim3 g2(C_DIM / 32, H_DIM / 32, E_NUM);  // w2[e][H][C] -> w2t[e][C][H]
    transpose_bf16_kernel<<<g2, blk, 0, stream>>>(w2, w2t, H_DIM, C_DIM);
  }

  // 3) gating + routing lists (8 waves/block, 1 token per wave)
  gate_kernel<<<N_TOK / 8, 256, 0, stream>>>(x, wg, cnt, tokidx, tokw);

  // 4) fused expert FFN (263 KB dynamic LDS per WG)
  hipFuncSetAttribute((const void*)moe_ffn_kernel,
                      hipFuncAttributeMaxDynamicSharedMemorySize, SHMEM_BYTES);
  moe_ffn_kernel<<<E_NUM * TILES_PER_E, 256, SHMEM_BYTES, stream>>>(
      xbf, w1t, b1, w2t, b2, cnt, tokidx, tokw, out);
}
